// GRUExtendedKalmanBlock_57114475102795
// MI455X (gfx1250) — matrix operand order repaired and checked by reading, not compiled, not run on hardware
//
#include <hip/hip_runtime.h>

// ---------------------------------------------------------------------------
// GRU Extended-Kalman block, fused, for MI455X (gfx1250, wave32, WMMA bf16).
// Each wave processes a 16-row batch tile through the whole network.
// Weights (bf16, padded N%16, K%32) + biases (f32, padded N%16) staged in d_ws
// at constexpr offsets shared between host and device -> tiny kernarg struct.
// ---------------------------------------------------------------------------

typedef __bf16 bf16;
typedef __attribute__((ext_vector_type(16))) __bf16 v16bf;
typedef __attribute__((ext_vector_type(8)))  __bf16 v8bf;
typedef __attribute__((ext_vector_type(8)))  float  v8f;

union ABFrag { v16bf v; v8bf h[2]; };

#define WSYNC() asm volatile("s_wait_dscnt 0" ::: "memory")

// ---- layer table (order fixed; shared host/device) ------------------------
enum {
    L_QI0 = 0, L_QI1, L_QI2, L_QI3,
    L_QG_WIH, L_QG_WHH,
    L_PI0, L_PI1, L_PI2, L_PI3,
    L_PG_WIH, L_PG_WHH,
    L_PS0, L_PS1,
    L_SI0, L_SI1, L_SI2, L_SI3,
    L_SG_WIH, L_SG_WHH,
    L_KG0, L_KG1, L_KG2, L_KG3,
    L_SK0, L_SK1, L_SK2, L_SK3,
    L_PU0, L_PU1, L_PU2, L_PU3,
    N_LAYERS
};
// rows (N) and cols (K) of each weight matrix, and d_in index of W / bias.
constexpr int W_N[N_LAYERS] = {32,32,32,32, 768,768, 32,32,32,256, 768,768,
                               144,144, 24,24,24,144, 432,432,
                               192,192,192,192, 256,256,256,256, 256,256,256,256};
constexpr int W_K[N_LAYERS] = {16,32,32,32, 32,256, 16,32,32,32, 512,256,
                               256,144, 24,24,24,24, 288,144,
                               400,192,192,192, 336,256,256,256, 512,256,256,256};
constexpr int W_IDX[N_LAYERS] = {7,9,11,13, 15,16, 19,21,23,25, 27,28,
                                 31,33, 35,37,39,41, 43,44,
                                 47,49,51,53, 55,57,59,61, 63,65,67,69};
constexpr int B_IDX[N_LAYERS] = {8,10,12,14, 17,18, 20,22,24,26, 29,30,
                                 32,34, 36,38,40,42, 45,46,
                                 48,50,52,54, 56,58,60,62, 64,66,68,70};

constexpr int padN(int n) { return (n + 15) & ~15; }
constexpr int padK(int k) { return (k + 31) & ~31; }

struct OffTab { size_t w[N_LAYERS + 1]; size_t b[N_LAYERS + 1]; };
constexpr OffTab make_offs() {
    OffTab t{};
    size_t aw = 0, ab = 0;
    for (int i = 0; i < N_LAYERS; ++i) {
        t.w[i] = aw; aw += (size_t)padN(W_N[i]) * padK(W_K[i]);
        t.b[i] = ab; ab += (size_t)padN(W_N[i]);
    }
    t.w[N_LAYERS] = aw; t.b[N_LAYERS] = ab;
    return t;
}
constexpr OffTab OFF = make_offs();

static __device__ __forceinline__ float sigmoidf_(float x) {
    return 1.0f / (1.0f + __expf(-x));
}
static __device__ __forceinline__ float reluf_(float x) {
    return __builtin_amdgcn_fmed3f(x, 0.0f, __builtin_inff());   // v_med3
}

// ---- fragment loads -------------------------------------------------------
// A (activations, 16x32 bf16 tile from LDS, row-major, row stride `sstride`):
// lanes 0-15: M=lane, K={k0..k0+7, k0+16..k0+23}
// lanes 16-31: M=lane-16, K={k0+8..k0+15, k0+24..k0+31}
static __device__ __forceinline__ v16bf load_afrag(const bf16* buf, int sstride,
                                                   int k0, int lane) {
    const int half = lane >> 4;
    const int m    = lane & 15;
    const bf16* row = buf + m * sstride + k0 + half * 8;
    ABFrag t;
    t.h[0] = *(const v8bf*)(const void*)(row);
    t.h[1] = *(const v8bf*)(const void*)(row + 16);
    return t.v;
}
// B (weights [Npad x Kpad] bf16 row-major; B[k][n] = W[n][k]):
// lanes 0-15: N=lane, K={k0..k0+7, k0+16..k0+23} ; lanes 16-31: N=lane-16, K={k0+8..k0+15, k0+24..k0+31}
static __device__ __forceinline__ v16bf load_bfrag(const bf16* W, int Kp,
                                                   int n0, int k0, int lane) {
    const int half = lane >> 4;
    const int n    = lane & 15;
    const bf16* p = W + (size_t)(n0 + n) * Kp + k0 + half * 8;
    ABFrag t;
    t.h[0] = *(const v8bf*)(const void*)(p);
    t.h[1] = *(const v8bf*)(const void*)(p + 16);
    return t.v;
}
static __device__ __forceinline__ v8f mm_acc(v8f acc, const bf16* src, int sstride,
                                             int Kp, const bf16* W, int n0, int lane) {
    for (int k0 = 0; k0 < Kp; k0 += 32) {
        v16bf a = load_afrag(src, sstride, k0, lane);
        v16bf b = load_bfrag(W, Kp, n0, k0, lane);
        acc = __builtin_amdgcn_wmma_f32_16x16x32_bf16(false, a, false, b,
                                                      (short)0, acc, false, false);
    }
    return acc;
}

// ---- small LDS utilities (per-wave, 16 rows) ------------------------------
static __device__ __forceinline__ void load_g2l(bf16* dst, int dstride, int dcol0,
                                                const float* src, int srcW, int ncols,
                                                long r0, int lane) {
    for (int i = lane; i < 16 * ncols; i += 32) {
        int r = i / ncols, c = i - r * ncols;
        dst[r * dstride + dcol0 + c] = (bf16)src[(size_t)(r0 + r) * srcW + c];
    }
}
static __device__ __forceinline__ void zero_cols(bf16* dst, int dstride,
                                                 int c0, int nc, int lane) {
    for (int i = lane; i < 16 * nc; i += 32) {
        int r = i / nc, c = i - r * nc;
        dst[r * dstride + c0 + c] = (bf16)0.0f;
    }
}
static __device__ __forceinline__ void copy_cols(bf16* dst, int dstride, int dc0,
                                                 const bf16* src, int sstride, int sc0,
                                                 int nc, int lane) {
    for (int i = lane; i < 16 * nc; i += 32) {
        int r = i / nc, c = i - r * nc;
        dst[r * dstride + dc0 + c] = src[r * sstride + sc0 + c];
    }
}

// ---- Linear layer: dst = act(src @ W^T + b) -------------------------------
template <bool RELU, bool HASD, bool HASG>
static __device__ __forceinline__
void linear_layer(const bf16* W, const float* bias, int Nv, int Kp,
                  const bf16* src, int sstride,
                  bf16* dst, int dstride, int dcol0,
                  float* gout, int goutW, long r0, int lane) {
    const int n  = lane & 15;
    const int mb = (lane >> 4) * 8;
    const int Npad = (Nv + 15) & ~15;
    for (int n0 = 0; n0 < Npad; n0 += 16) {
        const float bv = bias[n0 + n];            // bias zero-padded to Npad
        v8f acc;
#pragma unroll
        for (int v = 0; v < 8; ++v) acc[v] = bv;
        acc = mm_acc(acc, src, sstride, Kp, W, n0, lane);
#pragma unroll
        for (int v = 0; v < 8; ++v) {
            float x = acc[v];
            if constexpr (RELU) x = reluf_(x);
            if constexpr (HASD) dst[(mb + v) * dstride + dcol0 + n0 + n] = (bf16)x;
            if constexpr (HASG) gout[(size_t)(r0 + mb + v) * goutW + n0 + n] = x;
        }
    }
}

// ---- GRU cell: h' = (1-z)*tanh(in + r*hn) + z*h ---------------------------
// Gate row blocks in Wih/Whh: [r | z | n], each dh rows. dh % 16 == 0.
template <bool HASD, bool HASG>
static __device__ __forceinline__
void gru_cell(const bf16* Wih, int Kxp, const bf16* Whh, int Khp,
              const float* bih, const float* bhh, int dh,
              const bf16* X, int xstride, const bf16* H, int hstride,
              bf16* dstH, int dstride, int dcol0,
              float* gout, int goutW, long r0, int lane) {
    const int n  = lane & 15;
    const int mb = (lane >> 4) * 8;
    for (int n0 = 0; n0 < dh; n0 += 16) {
        const int col = n0 + n;
        v8f ar, az, ai, ah;
        const float br = bih[col] + bhh[col];
        const float bz = bih[dh + col] + bhh[dh + col];
        const float bi = bih[2 * dh + col];
        const float bn = bhh[2 * dh + col];
#pragma unroll
        for (int v = 0; v < 8; ++v) { ar[v] = br; az[v] = bz; ai[v] = bi; ah[v] = bn; }

        ar = mm_acc(ar, X, xstride, Kxp, Wih, n0, lane);
        ar = mm_acc(ar, H, hstride, Khp, Whh, n0, lane);
        az = mm_acc(az, X, xstride, Kxp, Wih + (size_t)dh * Kxp, n0, lane);
        az = mm_acc(az, H, hstride, Khp, Whh + (size_t)dh * Khp, n0, lane);
        ai = mm_acc(ai, X, xstride, Kxp, Wih + (size_t)2 * dh * Kxp, n0, lane);
        ah = mm_acc(ah, H, hstride, Khp, Whh + (size_t)2 * dh * Khp, n0, lane);

#pragma unroll
        for (int v = 0; v < 8; ++v) {
            const float hprev = (float)H[(mb + v) * hstride + col];
            const float r  = sigmoidf_(ar[v]);
            const float z  = sigmoidf_(az[v]);
            const float nn = tanhf(ai[v] + r * ah[v]);
            const float hv = (1.0f - z) * nn + z * hprev;
            if constexpr (HASD) dstH[(mb + v) * dstride + dcol0 + col] = (bf16)hv;
            if constexpr (HASG) gout[(size_t)(r0 + mb + v) * goutW + col] = hv;
        }
    }
}

// ---------------------------------------------------------------------------
struct Params {
    const float *F1, *F2, *F3, *F4, *Qh, *Ph, *Sh;
    const bf16  *wb;    // all bf16 weights at OFF.w[]
    const float *bb;    // all f32 biases at OFF.b[]
    float *oKG, *oQ, *oP, *oS;
    int B;
};

// Per-wave LDS (bf16 elements): A 16x512, Bb 16x512, H 16x256, P 16x256, S 16x160
#define ELS_A   (16 * 512)
#define ELS_B   (16 * 512)
#define ELS_H   (16 * 256)
#define ELS_P   (16 * 256)
#define ELS_S   (16 * 160)
#define ELS_WAVE (ELS_A + ELS_B + ELS_H + ELS_P + ELS_S)   // 27136 (54272 B)
#define WAVES_PER_WG 4

__global__ __launch_bounds__(32 * WAVES_PER_WG, 1)
void ekf_fused_kernel(Params p) {
    extern __shared__ bf16 smem[];
    const int lane = threadIdx.x & 31;
    const int wave = threadIdx.x >> 5;
    const long r0 = ((long)blockIdx.x * WAVES_PER_WG + wave) * 16;
    if (r0 >= p.B) return;

    bf16* A  = smem + (size_t)wave * ELS_WAVE;
    bf16* Bb = A + ELS_A;
    bf16* H  = Bb + ELS_B;
    bf16* P  = H + ELS_H;
    bf16* S  = P + ELS_P;

    const bf16*  wb = p.wb;
    const float* bb = p.bb;
#define WGT(i) (wb + OFF.w[i])
#define BIA(i) (bb + OFF.b[i])

    // ---- QI(F4): 16->32->32->32->32, X_q ends in Bb[0..32) ----------------
    load_g2l(Bb, 512, 0, p.F4, 16, 16, r0, lane);
    zero_cols(Bb, 512, 16, 16, lane);                       WSYNC();
    linear_layer<1,1,0>(WGT(L_QI0), BIA(L_QI0), 32, 32, Bb, 512, A,  512, 0, nullptr, 0, r0, lane); WSYNC();
    linear_layer<1,1,0>(WGT(L_QI1), BIA(L_QI1), 32, 32, A,  512, Bb, 512, 0, nullptr, 0, r0, lane); WSYNC();
    linear_layer<1,1,0>(WGT(L_QI2), BIA(L_QI2), 32, 32, Bb, 512, A,  512, 0, nullptr, 0, r0, lane); WSYNC();
    linear_layer<0,1,0>(WGT(L_QI3), BIA(L_QI3), 32, 32, A,  512, Bb, 512, 0, nullptr, 0, r0, lane); WSYNC();

    // ---- Q_GRU: dh=256. Q -> A[0..256) and global oQ ----------------------
    load_g2l(H, 256, 0, p.Qh, 256, 256, r0, lane);          WSYNC();
    gru_cell<1,1>(WGT(L_QG_WIH), 32, WGT(L_QG_WHH), 256, BIA(L_QG_WIH), BIA(L_QG_WHH), 256,
                  Bb, 512, H, 256, A, 512, 0, p.oQ, 256, r0, lane); WSYNC();

    // ---- PI(F3): 16->32->32->32->256, P_inp -> A[256..512) ----------------
    load_g2l(Bb, 512, 64, p.F3, 16, 16, r0, lane);
    zero_cols(Bb, 512, 80, 16, lane);                       WSYNC();
    linear_layer<1,1,0>(WGT(L_PI0), BIA(L_PI0), 32, 32, Bb + 64, 512, Bb, 512, 0,  nullptr, 0, r0, lane); WSYNC();
    linear_layer<1,1,0>(WGT(L_PI1), BIA(L_PI1), 32, 32, Bb,      512, Bb, 512, 32, nullptr, 0, r0, lane); WSYNC();
    linear_layer<1,1,0>(WGT(L_PI2), BIA(L_PI2), 32, 32, Bb + 32, 512, Bb, 512, 0,  nullptr, 0, r0, lane); WSYNC();
    linear_layer<0,1,0>(WGT(L_PI3), BIA(L_PI3), 256, 32, Bb,     512, A,  512, 256, nullptr, 0, r0, lane); WSYNC();

    // ---- P_GRU: X = concat(Q, P_inp) = A[0..512), dh=256 -> Pbuf ----------
    load_g2l(H, 256, 0, p.Ph, 256, 256, r0, lane);          WSYNC();
    gru_cell<1,0>(WGT(L_PG_WIH), 512, WGT(L_PG_WHH), 256, BIA(L_PG_WIH), BIA(L_PG_WHH), 256,
                  A, 512, H, 256, P, 256, 0, nullptr, 0, r0, lane); WSYNC();

    // ---- SI(concat(F1,F2)): 24->24->24->24->144, S_inp -> A[0..144) -------
    load_g2l(Bb, 512, 0,  p.F1, 12, 12, r0, lane);
    load_g2l(Bb, 512, 12, p.F2, 12, 12, r0, lane);
    zero_cols(Bb, 512, 24, 8, lane);                        WSYNC();
    linear_layer<1,1,0>(WGT(L_SI0), BIA(L_SI0), 24, 32, Bb,      512, Bb, 512, 32, nullptr, 0, r0, lane); WSYNC();
    linear_layer<1,1,0>(WGT(L_SI1), BIA(L_SI1), 24, 32, Bb + 32, 512, Bb, 512, 0,  nullptr, 0, r0, lane); WSYNC();
    linear_layer<1,1,0>(WGT(L_SI2), BIA(L_SI2), 24, 32, Bb,      512, Bb, 512, 32, nullptr, 0, r0, lane); WSYNC();
    linear_layer<0,1,0>(WGT(L_SI3), BIA(L_SI3), 144, 32, Bb + 32, 512, A, 512, 0,  nullptr, 0, r0, lane); WSYNC();

    // ---- PS(P): 256->144->144 -> A[144..288) ------------------------------
    linear_layer<1,1,0>(WGT(L_PS0), BIA(L_PS0), 144, 256, P, 256, Bb, 512, 0, nullptr, 0, r0, lane); WSYNC();
    zero_cols(Bb, 512, 144, 16, lane);                      WSYNC();
    linear_layer<0,1,0>(WGT(L_PS1), BIA(L_PS1), 144, 160, Bb, 512, A, 512, 144, nullptr, 0, r0, lane); WSYNC();

    // ---- S_GRU: X = concat(S_inp, PS(P)) = A[0..288), dh=144 --------------
    load_g2l(H, 256, 0, p.Sh, 144, 144, r0, lane);
    zero_cols(H, 256, 144, 16, lane);                       WSYNC();
    gru_cell<1,1>(WGT(L_SG_WIH), 288, WGT(L_SG_WHH), 160, BIA(L_SG_WIH), BIA(L_SG_WHH), 144,
                  A, 512, H, 256, S, 160, 0, p.oS, 144, r0, lane); WSYNC();
    zero_cols(S, 160, 144, 16, lane);                       WSYNC();

    // ---- KG(concat(S,P)): 400->192 x4. KG -> A[144..336) + global ---------
    copy_cols(A, 512, 0,   S, 160, 0, 144, lane);
    copy_cols(A, 512, 144, P, 256, 0, 256, lane);
    zero_cols(A, 512, 400, 16, lane);                       WSYNC();
    linear_layer<1,1,0>(WGT(L_KG0), BIA(L_KG0), 192, 416, A,        512, Bb, 512, 0,   nullptr, 0, r0, lane); WSYNC();
    linear_layer<1,1,0>(WGT(L_KG1), BIA(L_KG1), 192, 192, Bb,       512, Bb, 512, 192, nullptr, 0, r0, lane); WSYNC();
    linear_layer<1,1,0>(WGT(L_KG2), BIA(L_KG2), 192, 192, Bb + 192, 512, Bb, 512, 0,   nullptr, 0, r0, lane); WSYNC();
    linear_layer<0,1,1>(WGT(L_KG3), BIA(L_KG3), 192, 192, Bb,       512, A,  512, 144, p.oKG, 192, r0, lane); WSYNC();

    // ---- SK(concat(S,KG)): 336->256 x4 -> A[256..512) ---------------------
    copy_cols(A, 512, 0, S, 160, 0, 144, lane);
    zero_cols(A, 512, 336, 16, lane);                       WSYNC();
    linear_layer<1,1,0>(WGT(L_SK0), BIA(L_SK0), 256, 352, A,        512, Bb, 512, 0,   nullptr, 0, r0, lane); WSYNC();
    linear_layer<1,1,0>(WGT(L_SK1), BIA(L_SK1), 256, 256, Bb,       512, Bb, 512, 256, nullptr, 0, r0, lane); WSYNC();
    linear_layer<1,1,0>(WGT(L_SK2), BIA(L_SK2), 256, 256, Bb + 256, 512, Bb, 512, 0,   nullptr, 0, r0, lane); WSYNC();
    linear_layer<0,1,0>(WGT(L_SK3), BIA(L_SK3), 256, 256, Bb,       512, A,  512, 256, nullptr, 0, r0, lane); WSYNC();

    // ---- PU(concat(P,SK)): 512->256 x4 -> global oP -----------------------
    copy_cols(A, 512, 0, P, 256, 0, 256, lane);             WSYNC();
    linear_layer<1,1,0>(WGT(L_PU0), BIA(L_PU0), 256, 512, A,        512, Bb, 512, 0,   nullptr, 0, r0, lane); WSYNC();
    linear_layer<1,1,0>(WGT(L_PU1), BIA(L_PU1), 256, 256, Bb,       512, Bb, 512, 256, nullptr, 0, r0, lane); WSYNC();
    linear_layer<1,1,0>(WGT(L_PU2), BIA(L_PU2), 256, 256, Bb + 256, 512, Bb, 512, 0,   nullptr, 0, r0, lane); WSYNC();
    linear_layer<0,0,1>(WGT(L_PU3), BIA(L_PU3), 256, 256, Bb,       512, nullptr, 0, 0, p.oP, 256, r0, lane); WSYNC();
#undef WGT
#undef BIA
}

// ---- staging kernels ------------------------------------------------------
__global__ void convert_weight_kernel(const float* __restrict__ src, bf16* __restrict__ dst,
                                      int Nv, int Kv, int Np, int Kp) {
    const int total = Np * Kp;
    for (int i = blockIdx.x * blockDim.x + threadIdx.x; i < total; i += gridDim.x * blockDim.x) {
        const int nrow = i / Kp, k = i - nrow * Kp;
        float v = (nrow < Nv && k < Kv) ? src[(size_t)nrow * Kv + k] : 0.0f;
        dst[i] = (bf16)v;
    }
}
__global__ void copy_bias_kernel(const float* __restrict__ src, float* __restrict__ dst,
                                 int Nv, int Np) {
    for (int i = blockIdx.x * blockDim.x + threadIdx.x; i < Np; i += gridDim.x * blockDim.x)
        dst[i] = (i < Nv) ? src[i] : 0.0f;
}

// ---------------------------------------------------------------------------
extern "C" void kernel_launch(void* const* d_in, const int* in_sizes, int n_in,
                              void* d_out, int out_size, void* d_ws, size_t ws_size,
                              hipStream_t stream) {
    (void)n_in; (void)out_size; (void)ws_size;

    bf16*  wbase = (bf16*)d_ws;
    float* bbase = (float*)(wbase + OFF.w[N_LAYERS]);   // bias region after weights

    for (int i = 0; i < N_LAYERS; ++i) {
        const int Np = padN(W_N[i]), Kp = padK(W_K[i]);
        const int total = Np * Kp;
        convert_weight_kernel<<<(total + 255) / 256, 256, 0, stream>>>(
            (const float*)d_in[W_IDX[i]], wbase + OFF.w[i], W_N[i], W_K[i], Np, Kp);
        copy_bias_kernel<<<(Np + 255) / 256, 256, 0, stream>>>(
            (const float*)d_in[B_IDX[i]], bbase + OFF.b[i], W_N[i], Np);
    }

    const int Bn = in_sizes[0] / 12;                 // B from F1 (B x 12)
    Params p;
    p.F1 = (const float*)d_in[0]; p.F2 = (const float*)d_in[1];
    p.F3 = (const float*)d_in[2]; p.F4 = (const float*)d_in[3];
    p.Qh = (const float*)d_in[4]; p.Ph = (const float*)d_in[5];
    p.Sh = (const float*)d_in[6];
    p.wb = wbase; p.bb = bbase;
    float* out = (float*)d_out;
    p.oKG = out;
    p.oQ  = p.oKG + (size_t)Bn * 192;
    p.oP  = p.oQ  + (size_t)Bn * 256;
    p.oS  = p.oP  + (size_t)Bn * 256;
    p.B = Bn;

    const size_t shmem = (size_t)WAVES_PER_WG * ELS_WAVE * sizeof(bf16); // ~212 KB of 320 KB/WGP
    const int rows_per_wg = 16 * WAVES_PER_WG;
    const int grid = (Bn + rows_per_wg - 1) / rows_per_wg;
    ekf_fused_kernel<<<grid, 32 * WAVES_PER_WG, shmem, stream>>>(p);
}
